// TrajectoryTransformer_89970974916874
// MI455X (gfx1250) — matrix operand order repaired and checked by reading, not compiled, not run on hardware
//
#include <hip/hip_runtime.h>
#include <hip/hip_bf16.h>

// ---------------------------------------------------------------------------
// Types / helpers for CDNA5 WMMA (gfx1250, wave32)
// ---------------------------------------------------------------------------
typedef __attribute__((ext_vector_type(16))) __bf16 v16bf;
typedef __attribute__((ext_vector_type(8)))  float  v8f;

__device__ __forceinline__ unsigned bfbits(float f) {
  unsigned u = __builtin_bit_cast(unsigned, f);
  u += 0x7FFFu + ((u >> 16) & 1u);           // round-to-nearest-even
  return u >> 16;
}
__device__ __forceinline__ unsigned pack2bf(float lo, float hi) {
  return bfbits(lo) | (bfbits(hi) << 16);
}
__device__ __forceinline__ __bf16 f2bf(float f) {
  unsigned short h = (unsigned short)bfbits(f);
  return __builtin_bit_cast(__bf16, h);
}

__device__ __forceinline__ v8f wmma_bf16(v16bf a, v16bf b, v8f c) {
  // D = A(16x32) * B(32x16) + C(16x16), f32 accumulate
  return __builtin_amdgcn_wmma_f32_16x16x32_bf16(
      /*neg_a=*/false, a, /*neg_b=*/false, b,
      /*c_mod=*/(short)0, c, /*reuse_a=*/false, /*reuse_b=*/false);
}

// Model constants
#define D_MODEL 512
#define D_QKV   64
#define N_HEADS 8
#define D_FF    2048
#define BB      4
#define LE      1024
#define LD      256
#define VOUT    4096

// ---------------------------------------------------------------------------
// GEMM: C[M,N] = A[M,K] @ B[K,N] (+bias) (optional relu), fp32 in/out,
// bf16 WMMA compute. Block tile 128x128, 8 waves as 4(M)x2(N),
// wave tile 32x64 -> 8 WMMAs per 32-wide k-step.
// Requires M%128==0, N%128==0, K%32==0 (true for all uses here).
// ---------------------------------------------------------------------------
#define BM 128
#define BN 128
#define BK 32

__global__ __launch_bounds__(256) void gemm_bf16_wmma(
    const float* __restrict__ A, const float* __restrict__ B,
    const float* __restrict__ bias, float* __restrict__ C,
    int M, int N, int K, int relu)
{
  __shared__ __bf16 As[BM * BK];   // [m][k]   8 KB
  __shared__ __bf16 Bs[BN * BK];   // [n][k]   8 KB (transposed for contiguous frags)

  const int tid  = threadIdx.x;
  const int lane = tid & 31;
  const int wave = tid >> 5;       // 0..7
  const int wm   = wave & 3;       // wave row (0..3), 32 rows each
  const int wn   = wave >> 2;      // wave col (0..1), 64 cols each
  const int m0   = blockIdx.y * BM;
  const int n0   = blockIdx.x * BN;

  v8f acc[2][4] = {};

  for (int k0 = 0; k0 < K; k0 += BK) {
    // Prefetch next k-step lines (global_prefetch_b8)
    if (k0 + BK < K) {
      __builtin_prefetch(A + (size_t)(m0 + (tid >> 3)) * K + (k0 + BK) + (tid & 7) * 4, 0, 3);
      __builtin_prefetch(B + (size_t)(k0 + BK + ((tid >> 7) * 4)) * N + n0 + (tid & 127), 0, 3);
    }
    // A tile: 128x32 fp32 -> bf16 LDS. float4 loads, packed b64 stores.
    #pragma unroll
    for (int e = 0; e < (BM * BK / 4) / 256; ++e) {   // 4 slots/thread
      int slot = tid + e * 256;                       // 0..1023
      int r  = slot >> 3;                             // row (8 float4 per row)
      int c4 = (slot & 7) * 4;                        // k offset
      const float4 v = *(const float4*)(A + (size_t)(m0 + r) * K + k0 + c4);
      uint2 p; p.x = pack2bf(v.x, v.y); p.y = pack2bf(v.z, v.w);
      *(uint2*)(&As[r * BK + c4]) = p;
    }
    // B tile: 32x128 fp32 -> bf16 LDS transposed [n][k].
    // 4 strided scalar loads along k, one packed b64 store per slot.
    #pragma unroll
    for (int e = 0; e < (BN * BK / 4) / 256; ++e) {
      int slot = tid + e * 256;                       // 0..1023
      int kk = (slot >> 7) * 4;                       // k group
      int n  = slot & 127;
      const float* bp = B + (size_t)(k0 + kk) * N + n0 + n;
      float b0 = bp[0];
      float b1 = bp[(size_t)N];
      float b2 = bp[(size_t)2 * N];
      float b3 = bp[(size_t)3 * N];
      uint2 p; p.x = pack2bf(b0, b1); p.y = pack2bf(b2, b3);
      *(uint2*)(&Bs[n * BK + kk]) = p;
    }
    __syncthreads();

    // A fragments (16x32 bf16 layout): lane<16 -> K 0..7 & 16..23 of row lane;
    // lane>=16 -> K 8..15 & 24..31 of row lane-16.
    v16bf af[2];
    #pragma unroll
    for (int mi = 0; mi < 2; ++mi) {
      int row = wm * 32 + mi * 16 + (lane & 15);
      int klo = (lane >> 4) * 8;
      #pragma unroll
      for (int i = 0; i < 8; ++i) af[mi][i]     = As[row * BK + klo + i];
      #pragma unroll
      for (int i = 0; i < 8; ++i) af[mi][8 + i] = As[row * BK + 16 + klo + i];
    }
    #pragma unroll
    for (int j = 0; j < 4; ++j) {
      // B fragment (32x16): lane<16 -> K 0..7 & 16..23 of col lane; lane>=16 -> K 8..15 & 24..31
      v16bf bfrag;
      int n    = wn * 64 + j * 16 + (lane & 15);
      int koff = (lane >> 4) * 8;
      #pragma unroll
      for (int i = 0; i < 16; ++i) bfrag[i] = Bs[n * BK + koff + (i & 7) + ((i & 8) << 1)];
      #pragma unroll
      for (int mi = 0; mi < 2; ++mi)
        acc[mi][j] = wmma_bf16(af[mi], bfrag, acc[mi][j]);
    }
    __syncthreads();
  }

  // Epilogue: C layout -> row = vgpr + 8*(lane>=16), col = lane%16
  #pragma unroll
  for (int mi = 0; mi < 2; ++mi) {
    const int rbase = m0 + wm * 32 + mi * 16 + 8 * (lane >> 4);
    #pragma unroll
    for (int j = 0; j < 4; ++j) {
      int col = n0 + wn * 64 + j * 16 + (lane & 15);
      float bv = bias ? bias[col] : 0.0f;
      #pragma unroll
      for (int i = 0; i < 8; ++i) {
        float v = acc[mi][j][i] + bv;
        if (relu) v = fmaxf(v, 0.0f);
        C[(size_t)(rbase + i) * N + col] = v;
      }
    }
  }
}

// ---------------------------------------------------------------------------
// Flash-style attention: one wave per (b, head, 16-query tile).
// Q,K,V layout: [b][pos][h*64+d] fp32. Output same layout (heads concat).
// mask: int (B, Ls) "enc_mask" style, or causal (s > l masked) if causal!=0.
// Score tile 16(l) x 32(s) via 4 WMMAs, online softmax, P@V via 4 WMMAs.
// ---------------------------------------------------------------------------
__global__ __launch_bounds__(32) void attention_wmma(
    const float* __restrict__ Q, const float* __restrict__ Kc,
    const float* __restrict__ Vc, float* __restrict__ O,
    const int* __restrict__ mask, int Lq, int Ls, int causal)
{
  __shared__ __bf16 Ks[32 * 64];  // [s][d]
  __shared__ __bf16 Vt[64 * 32];  // [d][s]
  __shared__ __bf16 Ps[16 * 32];  // [l][s]

  const int lane = threadIdx.x;
  const int l0   = blockIdx.x * 16;
  const int h    = blockIdx.y;
  const int b    = blockIdx.z;
  const size_t qbase = (size_t)b * Lq * D_MODEL + h * D_QKV;
  const size_t kbase = (size_t)b * Ls * D_MODEL + h * D_QKV;

  // Q fragments for the two 32-wide d-chunks (d = 0..63)
  v16bf qf[2];
  {
    int row = lane & 15;
    int klo = (lane >> 4) * 8;
    const float* qrow = Q + qbase + (size_t)(l0 + row) * D_MODEL;
    #pragma unroll
    for (int ks = 0; ks < 2; ++ks) {
      #pragma unroll
      for (int i = 0; i < 8; ++i) qf[ks][i]     = f2bf(qrow[ks * 32 + klo + i]);
      #pragma unroll
      for (int i = 0; i < 8; ++i) qf[ks][8 + i] = f2bf(qrow[ks * 32 + 16 + klo + i]);
    }
  }

  v8f o[4] = {};
  float mrow[8], lrow[8];
  #pragma unroll
  for (int i = 0; i < 8; ++i) { mrow[i] = -1e30f; lrow[i] = 0.0f; }
  const float scale = 0.125f;   // 1/sqrt(64)

  for (int sc = 0; sc < Ls; sc += 32) {
    // K chunk [32][64]: float4 loads, packed b64 stores (512 slots / 32 lanes)
    #pragma unroll
    for (int e = 0; e < 16; ++e) {
      int slot = lane + e * 32;
      int s  = slot >> 4;
      int d4 = (slot & 15) * 4;
      const float4 kv = *(const float4*)(Kc + kbase + (size_t)(sc + s) * D_MODEL + d4);
      uint2 p; p.x = pack2bf(kv.x, kv.y); p.y = pack2bf(kv.z, kv.w);
      *(uint2*)(&Ks[s * 64 + d4]) = p;
    }
    // V chunk transposed [d][s]: 4 strided loads along s, packed b64 store
    #pragma unroll
    for (int e = 0; e < 16; ++e) {
      int slot = lane + e * 32;
      int d  = slot >> 3;
      int s4 = (slot & 7) * 4;
      const float* vp = Vc + kbase + (size_t)(sc + s4) * D_MODEL + d;
      float v0 = vp[0];
      float v1 = vp[D_MODEL];
      float v2 = vp[2 * D_MODEL];
      float v3 = vp[3 * D_MODEL];
      uint2 p; p.x = pack2bf(v0, v1); p.y = pack2bf(v2, v3);
      *(uint2*)(&Vt[d * 32 + s4]) = p;
    }
    __syncthreads();

    // Scores: S(16x32) = Q(16x64) @ K^T(64x32), two 16-col halves x two k-steps
    v8f st[2] = {};
    #pragma unroll
    for (int half = 0; half < 2; ++half) {
      #pragma unroll
      for (int ks = 0; ks < 2; ++ks) {
        v16bf bfrag;
        int s    = half * 16 + (lane & 15);
        int koff = ks * 32 + (lane >> 4) * 8;
        #pragma unroll
        for (int i = 0; i < 16; ++i) bfrag[i] = Ks[s * 64 + koff + (i & 7) + ((i & 8) << 1)];
        st[half] = wmma_bf16(qf[ks], bfrag, st[half]);
      }
    }

    // Mask + online softmax (row = vgpr_i + 8*(lane>=16), col = lane%16)
    float p[2][8];
    #pragma unroll
    for (int i = 0; i < 8; ++i) {
      int row = l0 + i + 8 * (lane >> 4);
      float mx = -1e30f;
      #pragma unroll
      for (int half = 0; half < 2; ++half) {
        int s = sc + half * 16 + (lane & 15);
        float v = st[half][i] * scale;
        bool dead = causal ? (s > row) : (mask && mask[b * Ls + s] == 0);
        if (dead) v = -1e9f;
        st[half][i] = v;
        mx = fmaxf(mx, v);
      }
      for (int m2 = 1; m2 < 16; m2 <<= 1) mx = fmaxf(mx, __shfl_xor(mx, m2, 32));
      float nm    = fmaxf(mrow[i], mx);
      float alpha = __expf(mrow[i] - nm);
      float rs = 0.0f;
      #pragma unroll
      for (int half = 0; half < 2; ++half) {
        float pv = __expf(st[half][i] - nm);
        p[half][i] = pv;
        rs += pv;
      }
      for (int m2 = 1; m2 < 16; m2 <<= 1) rs += __shfl_xor(rs, m2, 32);
      lrow[i] = lrow[i] * alpha + rs;
      mrow[i] = nm;
      #pragma unroll
      for (int j = 0; j < 4; ++j) o[j][i] *= alpha;
    }

    // Reshape P (C layout) -> A fragment layout through LDS
    #pragma unroll
    for (int i = 0; i < 8; ++i) {
      int row = i + 8 * (lane >> 4);
      #pragma unroll
      for (int half = 0; half < 2; ++half)
        Ps[row * 32 + half * 16 + (lane & 15)] = f2bf(p[half][i]);
    }
    __syncthreads();

    v16bf pfrag;
    {
      int row = lane & 15;
      int klo = (lane >> 4) * 8;
      #pragma unroll
      for (int i = 0; i < 8; ++i) pfrag[i]     = Ps[row * 32 + klo + i];
      #pragma unroll
      for (int i = 0; i < 8; ++i) pfrag[8 + i] = Ps[row * 32 + 16 + klo + i];
    }
    // O(16x64) += P(16x32) @ V(32x64)
    #pragma unroll
    for (int j = 0; j < 4; ++j) {
      v16bf bfrag;
      int d    = j * 16 + (lane & 15);
      int koff = (lane >> 4) * 8;
      #pragma unroll
      for (int i = 0; i < 16; ++i) bfrag[i] = Vt[d * 32 + koff + (i & 7) + ((i & 8) << 1)];
      o[j] = wmma_bf16(pfrag, bfrag, o[j]);
    }
    __syncthreads();
  }

  // Normalize by row sums, write [b][l][h*64+d]
  #pragma unroll
  for (int j = 0; j < 4; ++j) {
    int col = h * D_QKV + j * 16 + (lane & 15);
    #pragma unroll
    for (int i = 0; i < 8; ++i) {
      int row = l0 + i + 8 * (lane >> 4);
      O[((size_t)b * Lq + row) * D_MODEL + col] = o[j][i] / lrow[i];
    }
  }
}

// ---------------------------------------------------------------------------
// Residual + LayerNorm (in place on X): X = LN(X + Y) * g + b, row = 512
// ---------------------------------------------------------------------------
__global__ __launch_bounds__(256) void residual_ln(
    float* __restrict__ X, const float* __restrict__ Y,
    const float* __restrict__ g, const float* __restrict__ bta)
{
  __shared__ float red0[8], red1[8];
  const int row = blockIdx.x;
  float* x = X + (size_t)row * D_MODEL;
  const float* y = Y + (size_t)row * D_MODEL;
  const int tid = threadIdx.x;

  float v[2], s = 0.0f, s2 = 0.0f;
  #pragma unroll
  for (int i = 0; i < 2; ++i) {
    int d = tid + i * 256;
    v[i] = x[d] + y[d];
    s += v[i]; s2 += v[i] * v[i];
  }
  for (int m = 1; m < 32; m <<= 1) { s += __shfl_xor(s, m, 32); s2 += __shfl_xor(s2, m, 32); }
  if ((tid & 31) == 0) { red0[tid >> 5] = s; red1[tid >> 5] = s2; }
  __syncthreads();
  float ts = 0.0f, ts2 = 0.0f;
  #pragma unroll
  for (int w = 0; w < 8; ++w) { ts += red0[w]; ts2 += red1[w]; }
  float mu  = ts * (1.0f / D_MODEL);
  float var = ts2 * (1.0f / D_MODEL) - mu * mu;
  float inv = rsqrtf(var + 1e-5f);
  #pragma unroll
  for (int i = 0; i < 2; ++i) {
    int d = tid + i * 256;
    x[d] = (v[i] - mu) * inv * g[d] + bta[d];
  }
}

// ---------------------------------------------------------------------------
// Embedding gather * sqrt(d_model) + sinusoidal PE (positions start at 1)
// ---------------------------------------------------------------------------
__global__ __launch_bounds__(256) void embed_pe(
    const int* __restrict__ idx, const float* __restrict__ emb,
    float* __restrict__ out, int L)
{
  const int row = blockIdx.x;          // b*L + l
  const int l   = row % L;
  const int tok = idx[row];
  const int tid = threadIdx.x;
  const float sqrtd = 22.62741699796952f;  // sqrt(512)
  const float pos   = (float)(l + 1);
  #pragma unroll
  for (int i = 0; i < 2; ++i) {
    int d = tid + i * 256;
    int pair = d >> 1;
    float div = __powf(10000.0f, (float)(2 * pair) * (1.0f / D_MODEL));
    float ang = pos / div;
    float pe  = (d & 1) ? __cosf(ang) : __sinf(ang);
    out[(size_t)row * D_MODEL + d] = emb[(size_t)tok * D_MODEL + d] * sqrtd + pe;
  }
}

// ---------------------------------------------------------------------------
// Repack per-head projection weight (H,512,64) -> (512, H*64) row-major
// ---------------------------------------------------------------------------
__global__ __launch_bounds__(256) void repack_headw(
    const float* __restrict__ W, float* __restrict__ Wp)
{
  int e = blockIdx.x * 256 + threadIdx.x;
  if (e >= D_MODEL * D_MODEL) return;
  int m = e >> 9;          // /512
  int r = e & 511;         // h*64+d
  int hh = r >> 6, dd = r & 63;
  Wp[m * D_MODEL + r] = W[((size_t)hh * D_MODEL + m) * D_QKV + dd];
}

// ---------------------------------------------------------------------------
// Row-wise log-softmax in place (row length N=4096)
// ---------------------------------------------------------------------------
__global__ __launch_bounds__(256) void log_softmax_rows(float* __restrict__ X, int N)
{
  __shared__ float red[8];
  const int row = blockIdx.x;
  float* x = X + (size_t)row * N;
  const int tid = threadIdx.x;

  float mx = -1e30f;
  for (int d = tid; d < N; d += 256) mx = fmaxf(mx, x[d]);
  for (int m = 1; m < 32; m <<= 1) mx = fmaxf(mx, __shfl_xor(mx, m, 32));
  if ((tid & 31) == 0) red[tid >> 5] = mx;
  __syncthreads();
  float gm = -1e30f;
  #pragma unroll
  for (int w = 0; w < 8; ++w) gm = fmaxf(gm, red[w]);
  __syncthreads();

  float s = 0.0f;
  for (int d = tid; d < N; d += 256) s += __expf(x[d] - gm);
  for (int m = 1; m < 32; m <<= 1) s += __shfl_xor(s, m, 32);
  if ((tid & 31) == 0) red[tid >> 5] = s;
  __syncthreads();
  float ts = 0.0f;
  #pragma unroll
  for (int w = 0; w < 8; ++w) ts += red[w];
  float lse = gm + __logf(ts);
  for (int d = tid; d < N; d += 256) x[d] = x[d] - lse;
}

// ---------------------------------------------------------------------------
// Host orchestration
// ---------------------------------------------------------------------------
static inline void launch_gemm(const float* A, const float* B, const float* bias,
                               float* C, int M, int N, int K, int relu,
                               hipStream_t s) {
  dim3 grid(N / BN, M / BM);
  gemm_bf16_wmma<<<grid, 256, 0, s>>>(A, B, bias, C, M, N, K, relu);
}

extern "C" void kernel_launch(void* const* d_in, const int* in_sizes, int n_in,
                              void* d_out, int out_size, void* d_ws, size_t ws_size,
                              hipStream_t stream) {
  (void)in_sizes; (void)n_in; (void)out_size; (void)ws_size;
  auto F = [&](int i) { return (const float*)d_in[i]; };

  const int* enc_x    = (const int*)d_in[0];   // (B, LE)
  const int* dec_x    = (const int*)d_in[1];   // (B, LD)
  const int* enc_mask = (const int*)d_in[2];   // (B, 1, LE) -> (B, LE)
  // d_in[3] = dec_mask: causal tril, replaced by arithmetic causal condition.

  // params pytree leaf indices (dict keys sorted, lists in order):
  // dec layers: base 4 + 20*l
  //   +0 ff.W1 +1 ff.W2 +2 ff.b1 +3 ff.b2
  //   +4 ln1.b +5 ln1.g +6 ln2.b +7 ln2.g +8 ln3.b +9 ln3.g
  //   +10 satt.Wk +11 satt.Wo +12 satt.Wq +13 satt.Wv +14 satt.bo
  //   +15 xatt.Wk +16 xatt.Wo +17 xatt.Wq +18 xatt.Wv +19 xatt.bo
  // 84 emb_in, 85 emb_out
  // enc layers: base 86 + 13*l
  //   +0 att.Wk +1 att.Wo +2 att.Wq +3 att.Wv +4 att.bo
  //   +5 ff.W1 +6 ff.W2 +7 ff.b1 +8 ff.b2
  //   +9 ln1.b +10 ln1.g +11 ln2.b +12 ln2.g
  // 138 out_b, 139 out_w
  const int EMB_IN = 84, EMB_OUT = 85, OUT_B = 138, OUT_W = 139;

  // Workspace layout (floats)
  float* ws = (float*)d_ws;
  float* eb = ws;                    // enc stream   (B*LE*512 = 2,097,152)
  float* db = eb + 2097152;          // dec stream   (B*LD*512 =   524,288)
  float* qb = db + 524288;           // Q buffer     (2,097,152)
  float* kb = qb + 2097152;          // K buffer     (2,097,152)
  float* vb = kb + 2097152;          // V buffer     (2,097,152)
  float* t1 = vb + 2097152;          // big temp     (B*LE*2048 = 8,388,608)
  float* t2 = t1 + 8388608;          // temp         (2,097,152)
  float* wq = t2 + 2097152;          // repacked Wq  (262,144)
  float* wk = wq + 262144;           // repacked Wk
  float* wv = wk + 262144;           // repacked Wv

  const int RPK_BLOCKS = (D_MODEL * D_MODEL + 255) / 256;

  // Embeddings + positional encoding
  embed_pe<<<BB * LE, 256, 0, stream>>>(enc_x, F(EMB_IN),  eb, LE);
  embed_pe<<<BB * LD, 256, 0, stream>>>(dec_x, F(EMB_OUT), db, LD);

  // ---------------- Encoder ----------------
  for (int l = 0; l < 4; ++l) {
    const int base = 86 + 13 * l;
    repack_headw<<<RPK_BLOCKS, 256, 0, stream>>>(F(base + 2), wq);
    repack_headw<<<RPK_BLOCKS, 256, 0, stream>>>(F(base + 0), wk);
    repack_headw<<<RPK_BLOCKS, 256, 0, stream>>>(F(base + 3), wv);
    launch_gemm(eb, wq, nullptr, qb, BB * LE, 512, 512, 0, stream);
    launch_gemm(eb, wk, nullptr, kb, BB * LE, 512, 512, 0, stream);
    launch_gemm(eb, wv, nullptr, vb, BB * LE, 512, 512, 0, stream);
    attention_wmma<<<dim3(LE / 16, N_HEADS, BB), 32, 0, stream>>>(
        qb, kb, vb, t2, enc_mask, LE, LE, 0);
    launch_gemm(t2, F(base + 1), F(base + 4), t1, BB * LE, 512, 512, 0, stream);
    residual_ln<<<BB * LE, 256, 0, stream>>>(eb, t1, F(base + 10), F(base + 9));
    launch_gemm(eb, F(base + 5), F(base + 7), t1, BB * LE, D_FF, 512, 1, stream);
    launch_gemm(t1, F(base + 6), F(base + 8), t2, BB * LE, 512, D_FF, 0, stream);
    residual_ln<<<BB * LE, 256, 0, stream>>>(eb, t2, F(base + 12), F(base + 11));
  }

  // ---------------- Decoder ----------------
  for (int l = 0; l < 4; ++l) {
    const int base = 4 + 20 * l;
    // Self-attention (causal)
    repack_headw<<<RPK_BLOCKS, 256, 0, stream>>>(F(base + 12), wq);
    repack_headw<<<RPK_BLOCKS, 256, 0, stream>>>(F(base + 10), wk);
    repack_headw<<<RPK_BLOCKS, 256, 0, stream>>>(F(base + 13), wv);
    launch_gemm(db, wq, nullptr, qb, BB * LD, 512, 512, 0, stream);
    launch_gemm(db, wk, nullptr, kb, BB * LD, 512, 512, 0, stream);
    launch_gemm(db, wv, nullptr, vb, BB * LD, 512, 512, 0, stream);
    attention_wmma<<<dim3(LD / 16, N_HEADS, BB), 32, 0, stream>>>(
        qb, kb, vb, t2, nullptr, LD, LD, 1);
    launch_gemm(t2, F(base + 11), F(base + 14), t1, BB * LD, 512, 512, 0, stream);
    residual_ln<<<BB * LD, 256, 0, stream>>>(db, t1, F(base + 5), F(base + 4));
    // Cross-attention (K/V from final encoder stream)
    repack_headw<<<RPK_BLOCKS, 256, 0, stream>>>(F(base + 17), wq);
    repack_headw<<<RPK_BLOCKS, 256, 0, stream>>>(F(base + 15), wk);
    repack_headw<<<RPK_BLOCKS, 256, 0, stream>>>(F(base + 18), wv);
    launch_gemm(db, wq, nullptr, qb, BB * LD, 512, 512, 0, stream);
    launch_gemm(eb, wk, nullptr, kb, BB * LE, 512, 512, 0, stream);
    launch_gemm(eb, wv, nullptr, vb, BB * LE, 512, 512, 0, stream);
    attention_wmma<<<dim3(LD / 16, N_HEADS, BB), 32, 0, stream>>>(
        qb, kb, vb, t2, enc_mask, LD, LE, 0);
    launch_gemm(t2, F(base + 16), F(base + 19), t1, BB * LD, 512, 512, 0, stream);
    residual_ln<<<BB * LD, 256, 0, stream>>>(db, t1, F(base + 7), F(base + 6));
    // FFN
    launch_gemm(db, F(base + 0), F(base + 2), t1, BB * LD, D_FF, 512, 1, stream);
    launch_gemm(t1, F(base + 1), F(base + 3), t2, BB * LD, 512, D_FF, 0, stream);
    residual_ln<<<BB * LD, 256, 0, stream>>>(db, t2, F(base + 9), F(base + 8));
  }

  // ---------------- Output projection + log-softmax ----------------
  launch_gemm(db, F(OUT_W), F(OUT_B), (float*)d_out, BB * LD, VOUT, 512, 0, stream);
  log_softmax_rows<<<BB * LD, 256, 0, stream>>>((float*)d_out, VOUT);
}
